// TensorCircuit_82987358093981
// MI455X (gfx1250) — compile-verified, matrix operand order checked by reading, not run on hardware
//
#include <hip/hip_runtime.h>

// ---------------- problem constants (from reference) ----------------
#define V_   128
#define K_   32
#define C_   256
#define L_   4
#define N_   4096   // == V_*K_ : nm always has 4096 rows
#define F_   16
#define E_   8192
#define B_   512
#define EPSF 1e-6f

typedef _Float16 v16h __attribute__((ext_vector_type(16)));
typedef float    v8f  __attribute__((ext_vector_type(8)));

__device__ __forceinline__ float waveSum(float v) {
#pragma unroll
  for (int off = 16; off > 0; off >>= 1) v += __shfl_down(v, off, 32);
  return v;
}

// lse_in[vk] = logsumexp_c log(p+eps) = log( sum_c (p_c + eps) )
// one wave per row; grid 512 x block 256 (8 waves) covers 4096 rows
__global__ void k_lse(const float* __restrict__ ip, float* __restrict__ lse) {
  const int wave = threadIdx.x >> 5, lane = threadIdx.x & 31;
  const int row  = (blockIdx.x << 3) + wave;
  const float* p = ip + (size_t)row * C_;
  float s = 0.f;
  for (int i = lane; i < C_; i += 32) s += p[i] + EPSF;
  s = waveSum(s);
  if (lane == 0) lse[row] = __logf(s);
}

// normalized sum-weights in f16: w = (p+eps)/rowsum16  (exp of normalized log-weight, exactly)
__global__ void k_wsum(const float* __restrict__ sp, _Float16* __restrict__ wh) {
  const int idx = blockIdx.x * blockDim.x + threadIdx.x; // < L*N*F
  float v = sp[idx] + EPSF;
  float s = v;
#pragma unroll
  for (int off = 8; off > 0; off >>= 1) s += __shfl_xor(s, off, 16);
  wh[idx] = (_Float16)(v / s);
}

// normalized log root params: lr[n] = log(p+eps) - log(sum(p+eps))
__global__ void k_lr(const float* __restrict__ rp, float* __restrict__ lr) {
  __shared__ float red[8];
  const int t = threadIdx.x;
  float s = 0.f;
  for (int i = t; i < N_; i += 256) s += rp[i] + EPSF;
  s = waveSum(s);
  if ((t & 31) == 0) red[t >> 5] = s;
  __syncthreads();
  if (t == 0) {
    float tot = 0.f;
#pragma unroll
    for (int i = 0; i < 8; ++i) tot += red[i];
    red[0] = __logf(tot);
  }
  __syncthreads();
  const float lt = red[0];
  for (int i = t; i < N_; i += 256) lr[i] = __logf(rp[i] + EPSF) - lt;
}

// fused indices: fc[l,n,f,s] = prod_child_ids[l, sum_child_ids[l,n,f], s]
__global__ void k_fc(const int* __restrict__ pc, const int* __restrict__ sid,
                     int* __restrict__ fc) {
  const int idx = blockIdx.x * blockDim.x + threadIdx.x; // < L*N*F
  const int l = idx >> 16;                               // N*F = 65536
  const int e = sid[idx];
  const int* p = pc + (((size_t)l * E_ + e) << 1);
  fc[2 * idx]     = p[0];
  fc[2 * idx + 1] = p[1];
}

// leaf layer: nm0[vk,b] = log(ip[vk, x[b,v]] + eps) - lse_in[vk]
__global__ void k_nm0(const int* __restrict__ inp, const float* __restrict__ ip,
                      const float* __restrict__ lse, float* __restrict__ nm) {
  const int idx = blockIdx.x * blockDim.x + threadIdx.x; // < N_*B_
  const int b = idx & (B_ - 1);
  const int vk = idx >> 9;
  const int v = vk >> 5;                                 // K_ = 32
  const int x = inp[b * V_ + v];
  nm[idx] = __logf(ip[((size_t)vk << 8) + x] + EPSF) - lse[vk];
}

// ---------------- per-layer WMMA kernel ----------------
// block = 64 threads (2 waves), tile = 16 nodes x 32 batch cols.
// nm'[n,b] = m[n,b] + log( sum_f w[n,f] * exp(ch[n,f,b] - m[n,b]) )
// inner sum done as block-diagonal GEMM: A(16x256 f16, block-diag weights)
// x B(256x16 f16, exp-shifted children) via 8x v_wmma_f32_16x16x32_f16.
__global__ void __launch_bounds__(64) k_sum(const float* __restrict__ nm_in,
                                            float* __restrict__ nm_out,
                                            const int* __restrict__ fc_l,
                                            const _Float16* __restrict__ w_l) {
  __shared__ float chs[16][16][32];  // [node][f][b]  32 KB
  __shared__ float mrow[16][32];     // per-(n,b) max  2 KB
  __shared__ int   fcs[512];         // fused child ids 2 KB

  const int tid  = threadIdx.x;
  const int lane = tid & 31;
  const int wave = tid >> 5;
  const int ntile = blockIdx.x;       // 0..255 (node tile of 16)
  const int btile = blockIdx.y << 5;  // batch offset, step 32

  // stage fused child ids for this node tile
  const int* fsrc = fc_l + ((size_t)ntile << 9);
#pragma unroll
  for (int i = 0; i < 8; ++i) fcs[tid + 64 * i] = fsrc[tid + 64 * i];
  __syncthreads();

  // gather children: ch[n,f,b] = nm[c0,b] + nm[c1,b]  (coalesced over b)
  for (int i = tid; i < 16 * 16 * 32; i += 64) {
    const int b = i & 31, nf = i >> 5;
    const int c0 = fcs[2 * nf], c1 = fcs[2 * nf + 1];
    chs[nf >> 4][nf & 15][b] =
        nm_in[(size_t)c0 * B_ + btile + b] + nm_in[(size_t)c1 * B_ + btile + b];
  }
  __syncthreads();

  // per-(n,b) max over F=16 (shift for safe exp in f16)
  for (int i = tid; i < 16 * 32; i += 64) {
    const int b = i & 31, n = i >> 5;
    float m = chs[n][0][b];
#pragma unroll
    for (int f = 1; f < 16; ++f) m = fmaxf(m, chs[n][f][b]);
    mrow[n][b] = m;
  }
  __syncthreads();

  const int M    = lane & 15;        // row (A) / col (B,C/D) within tile
  const int hi   = lane >> 4;        // lane-half selects K sub-range
  const int bcol = (wave << 4) | M;  // this wave's batch column in the 32-tile

  // per-lane weight slice: wv[j] = w[M, hi*8 + j]  (constant across K-chunks)
  _Float16 wv[8];
  const _Float16* wrow = w_l + (((size_t)(ntile << 4) + M) << 4) + (hi << 3);
#pragma unroll
  for (int j = 0; j < 8; ++j) wv[j] = wrow[j];

  v8f acc = {0.f, 0.f, 0.f, 0.f, 0.f, 0.f, 0.f, 0.f};
#pragma unroll
  for (int kk = 0; kk < 8; ++kk) {   // K = 256 in chunks of 32; chunk kk covers nodes 2kk,2kk+1
    v16h a, bfrag;
    const bool lo = (M == 2 * kk);
    const bool hi2 = (M == 2 * kk + 1);
#pragma unroll
    for (int j = 0; j < 8; ++j) {    // ISA 16-bit A layout: j<8 -> K=hi*8+j, j>=8 -> K=16+hi*8+(j-8)
      a[j]     = lo  ? wv[j] : (_Float16)0.f;
      a[j + 8] = hi2 ? wv[j] : (_Float16)0.f;
    }
#pragma unroll
    for (int j = 0; j < 16; ++j) {   // B: K index -> (node, f) of the block-diag layout
      const int f  = (hi << 3) + (j & 7);
      const int nl = (kk << 1) + (j >> 3);
      bfrag[j] = (_Float16)__expf(chs[nl][f][bcol] - mrow[nl][bcol]);
    }
    acc = __builtin_amdgcn_wmma_f32_16x16x32_f16(false, a, false, bfrag,
                                                 (short)0, acc, false, false);
  }

  // C/D layout: VGPR r -> row M = r + 8*hi, col = lane&15 (within this wave's 16 cols)
#pragma unroll
  for (int r = 0; r < 8; ++r) {
    const int nrow = r + (hi << 3);
    const int n = (ntile << 4) + nrow;
    nm_out[(size_t)n * B_ + btile + bcol] = mrow[nrow][bcol] + __logf(acc[r]);
  }
}

// out[b] = logsumexp_n (nm[n,b] + lr[n]);  one block per batch element
__global__ void k_root(const float* __restrict__ nm, const float* __restrict__ lr,
                       float* __restrict__ out) {
  __shared__ float red[8];
  const int b = blockIdx.x, t = threadIdx.x;
  float m = -3.0e38f;
  for (int n = t; n < N_; n += 256) m = fmaxf(m, nm[(size_t)n * B_ + b] + lr[n]);
#pragma unroll
  for (int off = 16; off > 0; off >>= 1) m = fmaxf(m, __shfl_down(m, off, 32));
  if ((t & 31) == 0) red[t >> 5] = m;
  __syncthreads();
  if (t == 0) {
    float mm = red[0];
#pragma unroll
    for (int i = 1; i < 8; ++i) mm = fmaxf(mm, red[i]);
    red[0] = mm;
  }
  __syncthreads();
  const float gm = red[0];
  __syncthreads();
  float s = 0.f;
  for (int n = t; n < N_; n += 256) s += __expf(nm[(size_t)n * B_ + b] + lr[n] - gm);
  s = waveSum(s);
  if ((t & 31) == 0) red[t >> 5] = s;
  __syncthreads();
  if (t == 0) {
    float tot = 0.f;
#pragma unroll
    for (int i = 0; i < 8; ++i) tot += red[i];
    out[b] = gm + __logf(tot);
  }
}

extern "C" void kernel_launch(void* const* d_in, const int* in_sizes, int n_in,
                              void* d_out, int out_size, void* d_ws, size_t ws_size,
                              hipStream_t stream) {
  const int*   inp = (const int*)d_in[0];    // (B,V)
  const int*   pc  = (const int*)d_in[1];    // (L,E,2)
  const int*   sid = (const int*)d_in[2];    // (L,N,F)
  const float* ip  = (const float*)d_in[3];  // (V,K,C)
  const float* sp  = (const float*)d_in[4];  // (L,N,F)
  const float* rp  = (const float*)d_in[5];  // (N,)
  float* out = (float*)d_out;                // (B,)

  char* ws = (char*)d_ws;
  const size_t nmBytes = (size_t)N_ * B_ * sizeof(float);  // 8 MB each
  float*    nmA = (float*)(ws);
  float*    nmB = (float*)(ws + nmBytes);
  float*    lse = (float*)(ws + 2 * nmBytes);              // 4096 f32
  float*    lr  = lse + N_;                                // 4096 f32
  _Float16* wh  = (_Float16*)(lr + N_);                    // L*N*F f16 = 512 KB
  int*      fc  = (int*)((char*)wh + (size_t)L_ * N_ * F_ * sizeof(_Float16)); // 2 MB

  k_lse <<<N_ / 8,            256, 0, stream>>>(ip, lse);
  k_wsum<<<(L_*N_*F_) / 256,  256, 0, stream>>>(sp, wh);
  k_lr  <<<1,                 256, 0, stream>>>(rp, lr);
  k_fc  <<<(L_*N_*F_) / 256,  256, 0, stream>>>(pc, sid, fc);
  k_nm0 <<<(N_*B_) / 256,     256, 0, stream>>>(inp, ip, lse, nmA);

  float* src = nmA;
  float* dst = nmB;
  for (int l = 0; l < L_; ++l) {
    k_sum<<<dim3(N_ / 16, B_ / 32), 64, 0, stream>>>(
        src, dst, fc + 2 * (size_t)l * N_ * F_, wh + (size_t)l * N_ * F_);
    float* t = src; src = dst; dst = t;
  }
  k_root<<<B_, 256, 0, stream>>>(src, lr, out);
}
